// SequentialReader_8778913153081
// MI455X (gfx1250) — matrix operand order repaired and checked by reading, not compiled, not run on hardware
//
#include <hip/hip_runtime.h>
#include <hip/hip_bf16.h>

typedef __attribute__((ext_vector_type(16))) _Float16 v16h;
typedef __attribute__((ext_vector_type(8)))  float    v8f;

namespace {
constexpr int Hn = 10;   // hidden size
constexpr int Rn = 64;   // regions / timesteps
constexpr int Fn = 5;    // feature size

__device__ __forceinline__ float fast_rcp(float x) {
  return __builtin_amdgcn_rcpf(x);   // v_rcp_f32 (trans op), ~1 ulp
}
__device__ __forceinline__ float tanh_fast(float x) {
#if __has_builtin(__builtin_amdgcn_tanhf)
  return __builtin_amdgcn_tanhf(x);  // CDNA5 v_tanh_f32
#else
  // tanh(x) = 1 - 2/(exp(2x)+1); saturates correctly at +/-1
  return 1.0f - 2.0f * fast_rcp(__expf(2.0f * x) + 1.0f);
#endif
}
__device__ __forceinline__ float sigf(float x) {
#if __has_builtin(__builtin_amdgcn_tanhf)
  // sigmoid(x) = 0.5*tanh(x/2) + 0.5 : v_mul + v_tanh + v_fmac (one trans op)
  return __builtin_fmaf(0.5f, __builtin_amdgcn_tanhf(0.5f * x), 0.5f);
#else
  return fast_rcp(1.0f + __expf(-x));
#endif
}
} // namespace

// One wave32 owns 16 batch rows for the full 64-step recurrence.
// A operand packs [x_t | h] along K (K=0..9 x, K=10..19 h, rest 0), so a single
// v_wmma_f32_16x16x32_f16 computes x@W_ih.T + h@W_hh.T for one 16-wide gate tile.
__global__ __launch_bounds__(256, 1) void lstm_fused_wmma(
    const float* __restrict__ r_emb, const float* __restrict__ feature,
    const float* __restrict__ h0,    const float* __restrict__ c0,
    const float* __restrict__ W_ih,  const float* __restrict__ W_hh,
    const float* __restrict__ b_ih,  const float* __restrict__ b_hh,
    const float* __restrict__ Wr,    const float* __restrict__ br,
    const float* __restrict__ Wh,    const float* __restrict__ bh,
    const float* __restrict__ Wfc,   const float* __restrict__ bfc,
    float* __restrict__ out)
{
  __shared__ float wfc_s[Rn * Hn];   // 640 floats, block-shared
  __shared__ float wfc_tail[Fn];
  // B-operand staging: [mat][n][k], k contiguous so one lane's 16 K-values
  // (K = 8*hi..8*hi+7 and 16+8*hi..16+8*hi+7 at column n) are a single 32-byte LDS vector load.
  __shared__ __align__(32) _Float16 bmat_s[5][16][32];
  __shared__ float hbuf[8][16 * 17]; // per-wave h transpose tile (pad 17 vs bank conflicts)
  __shared__ float rowsum[8][16];

  const int tid  = threadIdx.x;
  const int wave = tid >> 5;
  const int lane = tid & 31;
  const int col  = lane & 15;   // N column / h index in C-layout
  const int hi   = lane >> 4;   // lane half
  const int base = (blockIdx.x * 8 + wave) * 16;   // first batch row of this wave

  for (int i = tid; i < Rn * Hn; i += 256) wfc_s[i] = Wfc[i];
  if (tid < Fn) wfc_tail[tid] = Wfc[Rn * Hn + tid];

  // Cooperative build of the 5 zero-padded stacked-weight operand images:
  // mats 0..3 = gate tiles i/f/g/o of [W_ih;W_hh]^T, mat 4 = [Wr;Wh]^T.
  for (int idx = tid; idx < 5 * 16 * 32; idx += 256) {
    const int k   = (idx & 7) | ((idx >> 1) & 8) | ((idx << 1) & 16);
    const int n   = (idx >> 5) & 15;
    const int mat = idx >> 9;
    float v = 0.0f;
    if (n < Hn) {
      if (mat < 4) {
        if (k < Hn)          v = W_ih[(mat * Hn + n) * Hn + k];
        else if (k < 2 * Hn) v = W_hh[(mat * Hn + n) * Hn + (k - Hn)];
      } else {
        if (k < Hn)          v = Wr[n * Hn + k];
        else if (k < 2 * Hn) v = Wh[n * Hn + (k - Hn)];
      }
    }
    (&bmat_s[0][0][0])[idx] = (_Float16)v;
  }
  __syncthreads();

  // ---- B operands (f16 32x16, K-major): element e of lane = K = 8*hi + e (e<8), 16 + 8*hi + (e-8) (e>=8), N = col
  v16h Bg[4];
  #pragma unroll
  for (int t = 0; t < 4; ++t)
    Bg[t] = *(const v16h*)&bmat_s[t][col][16 * hi];
  const v16h Bga = *(const v16h*)&bmat_s[4][col][16 * hi];

  float biasg[4];
  #pragma unroll
  for (int t = 0; t < 4; ++t)
    biasg[t] = (col < Hn) ? (b_ih[t * Hn + col] + b_hh[t * Hn + col]) : 0.0f;
  const float biasr = (col < Hn) ? (br[col] + bh[col]) : 0.0f;

  // ---- state in C-layout: row = r + 8*hi, col = lane%16 (cols >=10 inert zeros)
  v8f c;
  #pragma unroll
  for (int r = 0; r < 8; ++r)
    c[r] = (col < Hn) ? c0[(size_t)(base + r + 8 * hi) * Hn + col] : 0.0f;

  // ---- A operand, h slots preloaded from h0.
  //  lo lanes (row M=lane):   e0..7 = x[0..7],  e8..11 = h[6..9], e12..15 = 0
  //  hi lanes (row M=lane-16):e0=x[8],e1=x[9],  e2..7  = h[0..5], e8..15  = 0
  v16h A;
  #pragma unroll
  for (int e = 0; e < 16; ++e) A[e] = (_Float16)0.0f;
  if (!hi) {
    #pragma unroll
    for (int j = 0; j < 4; ++j)
      A[8 + j] = (_Float16)h0[(size_t)(base + lane) * Hn + 6 + j];
  } else {
    #pragma unroll
    for (int j = 0; j < 6; ++j)
      A[2 + j] = (_Float16)h0[(size_t)(base + lane - 16) * Hn + j];
  }

  v8f outacc;
  #pragma unroll
  for (int r = 0; r < 8; ++r) outacc[r] = 0.0f;

  float* hb = hbuf[wave];

  #pragma unroll 1
  for (int step = 0; step < Rn; ++step) {
    // x_t slice: each r_emb row streams sequentially (40B/step), cache friendly
    const float* xp = r_emb + ((size_t)(base + col) * Rn + step) * Hn;
    if (!hi) {
      #pragma unroll
      for (int e = 0; e < 8; ++e) A[e] = (_Float16)xp[e];
    } else {
      A[0] = (_Float16)xp[8];
      A[1] = (_Float16)xp[9];
    }
    if (step + 1 < Rn) __builtin_prefetch(xp + Hn, 0, 3);  // global_prefetch_b8

    // 4 gate tiles: D = [x|h] @ [W_ih;W_hh]^T + bias   (one WMMA per gate)
    v8f gi, gf, gg, go;
    #pragma unroll
    for (int r = 0; r < 8; ++r) {
      gi[r] = biasg[0]; gf[r] = biasg[1]; gg[r] = biasg[2]; go[r] = biasg[3];
    }
    gi = __builtin_amdgcn_wmma_f32_16x16x32_f16(false, A, false, Bg[0], (short)0, gi, false, false);
    gf = __builtin_amdgcn_wmma_f32_16x16x32_f16(false, A, false, Bg[1], (short)0, gf, false, false);
    gg = __builtin_amdgcn_wmma_f32_16x16x32_f16(false, A, false, Bg[2], (short)0, gg, false, false);
    go = __builtin_amdgcn_wmma_f32_16x16x32_f16(false, A, false, Bg[3], (short)0, go, false, false);

    // LSTM elementwise in C-layout (f32, single-trans-op activations)
    v8f hn;
    #pragma unroll
    for (int r = 0; r < 8; ++r) {
      const float cn = sigf(gf[r]) * c[r] + sigf(gi[r]) * tanh_fast(gg[r]);
      c[r]  = cn;
      hn[r] = sigf(go[r]) * tanh_fast(cn);
    }

    // Transpose h_new C-layout -> A-layout through per-wave LDS tile
    #pragma unroll
    for (int r = 0; r < 8; ++r) hb[(r + 8 * hi) * 17 + col] = hn[r];
    asm volatile("s_wait_dscnt 0" ::: "memory");
    if (!hi) {
      #pragma unroll
      for (int j = 0; j < 4; ++j) A[8 + j] = (_Float16)hb[lane * 17 + 6 + j];
    } else {
      #pragma unroll
      for (int j = 0; j < 6; ++j) A[2 + j] = (_Float16)hb[(lane - 16) * 17 + j];
    }

    // Output gate net: sigmoid([x|h_new] @ [Wr;Wh]^T + br + bh)  (one WMMA)
    v8f rg;
    #pragma unroll
    for (int r = 0; r < 8; ++r) rg[r] = biasr;
    rg = __builtin_amdgcn_wmma_f32_16x16x32_f16(false, A, false, Bga, (short)0, rg, false, false);

    // Fold final FC on the fly: out[b] += h_new*gate*Wfc[step*10+col]
    const float wv = (col < Hn) ? wfc_s[step * Hn + col] : 0.0f;
    #pragma unroll
    for (int r = 0; r < 8; ++r) outacc[r] += hn[r] * sigf(rg[r]) * wv;
  }

  // Reduce outacc across the 16 columns within each lane half
  #pragma unroll
  for (int r = 0; r < 8; ++r) {
    float s = outacc[r];
    s += __shfl_xor(s, 1, 32);
    s += __shfl_xor(s, 2, 32);
    s += __shfl_xor(s, 4, 32);
    s += __shfl_xor(s, 8, 32);
    if (col == 0) rowsum[wave][r + 8 * hi] = s;  // lanes 0 and 16
  }
  asm volatile("s_wait_dscnt 0" ::: "memory");

  if (lane < 16) {
    const int b = base + lane;
    float s = rowsum[wave][lane];
    float ft = 0.0f;
    #pragma unroll
    for (int f = 0; f < Fn; ++f) ft += feature[(size_t)b * Fn + f] * wfc_tail[f];
    out[b] = s + ft + bfc[0];
  }
}

extern "C" void kernel_launch(void* const* d_in, const int* in_sizes, int n_in,
                              void* d_out, int out_size, void* d_ws, size_t ws_size,
                              hipStream_t stream) {
  (void)n_in; (void)out_size; (void)d_ws; (void)ws_size;
  const float* r_emb   = (const float*)d_in[0];
  const float* feature = (const float*)d_in[1];
  const float* h0      = (const float*)d_in[2];
  const float* c0      = (const float*)d_in[3];
  const float* W_ih    = (const float*)d_in[4];
  const float* W_hh    = (const float*)d_in[5];
  const float* b_ih    = (const float*)d_in[6];
  const float* b_hh    = (const float*)d_in[7];
  const float* Wr      = (const float*)d_in[8];
  const float* br      = (const float*)d_in[9];
  const float* Wh      = (const float*)d_in[10];
  const float* bh      = (const float*)d_in[11];
  const float* Wfc     = (const float*)d_in[12];
  const float* bfc     = (const float*)d_in[13];
  float* out = (float*)d_out;

  const int B = in_sizes[0] / (Rn * Hn);   // 65536
  const int rows_per_block = 8 * 16;       // 8 waves x 16 batch rows
  dim3 grid((B + rows_per_block - 1) / rows_per_block);
  dim3 block(256);
  lstm_fused_wmma<<<grid, block, 0, stream>>>(
      r_emb, feature, h0, c0, W_ih, W_hh, b_ih, b_hh,
      Wr, br, Wh, bh, Wfc, bfc, out);
}
